// LocationSensitiveAttention_37323265802855
// MI455X (gfx1250) — matrix operand order repaired and checked by reading, not compiled, not run on hardware
//
#include <hip/hip_runtime.h>
#include <hip/hip_bf16.h>

#define B_    64
#define T_    1024
#define ATTN_ 128
#define RNN_  1024
#define ENC_  512
#define NFILT_ 32
#define KSZ_  31
#define PAD_  15

#define LDS_STRIDE 36   // shorts per row: keeps uint2 (b64) stores 8B-aligned

// LDS-only split barrier (CDNA5): waits DS ops only, leaves global loads
// (register prefetch of the next K-chunk) outstanding across the barrier.
#define LDS_BARRIER() asm volatile("s_wait_dscnt 0x0\n\t" \
                                   "s_barrier_signal -1\n\t" \
                                   "s_barrier_wait -1" ::: "memory")
// Hard machine-scheduler fence: nothing may be reordered across it, so all
// fragment ds_loads are issued (into distinct VGPRs) before the WMMA chain.
#define SCHED_FENCE() __builtin_amdgcn_sched_barrier(0)

typedef __attribute__((ext_vector_type(16))) __bf16       v16bf;
typedef __attribute__((ext_vector_type(8)))  float        v8f;
typedef __attribute__((ext_vector_type(8)))  unsigned int v8u;

__device__ __forceinline__ unsigned short f2bf(float x) {
    unsigned int u = __builtin_bit_cast(unsigned int, x);
    u += 0x7FFFu + ((u >> 16) & 1u);   // round-to-nearest-even
    return (unsigned short)(u >> 16);
}

// A fragment: 16x32 bf16 (ISA 7.12.2): lane m holds row M=m (both halves),
// VGPR v -> K = half*16 + 2v (contiguous order chosen here, not the ISA table order; matches B fragment)
__device__ __forceinline__ v16bf load_a_frag(const unsigned short* lA, int trow, int lane) {
    const int half = (lane >> 4) & 1, m = lane & 15;
    const unsigned short* base = lA + (trow + m) * LDS_STRIDE;
    v8u r;
#pragma unroll
    for (int v = 0; v < 8; ++v) {
        const int kb = (half << 4) + (v << 1);
        r[v] = *(const unsigned int*)(base + kb);
    }
    return __builtin_bit_cast(v16bf, r);
}

// B fragment: 32x16 bf16: column N = lane%16, lanes 0-15 hold K=0..15,
// lanes 16-31 K=16..31; staged [n][k] row-major so K pairs are contiguous.
__device__ __forceinline__ v16bf load_b_frag(const unsigned short* lB, int arow, int lane) {
    const int half = (lane >> 4) & 1, n = lane & 15;
    const unsigned short* base = lB + (arow + n) * LDS_STRIDE + (half << 4);
    v8u r;
#pragma unroll
    for (int v = 0; v < 8; ++v)
        r[v] = *(const unsigned int*)(base + (v << 1));
    return __builtin_bit_cast(v16bf, r);
}

// ---- tiny prep kernels ------------------------------------------------------

__global__ __launch_bounds__(256) void lsa_qproj_kernel(
        const float* __restrict__ query, const float* __restrict__ wq,
        float* __restrict__ qproj) {
    const int gid = blockIdx.x * 256 + threadIdx.x;      // 64*128
    const int b = gid >> 7, a = gid & 127;
    float acc = 0.f;
    for (int k = 0; k < RNN_; ++k)
        acc += query[b * RNN_ + k] * wq[a * RNN_ + k];
    qproj[gid] = acc;
}

// Wcomb[a][k] = sum_f W_loc_proj[a,f] * W_loc_conv[f,0,k]; zero-padded to k=32
__global__ __launch_bounds__(256) void lsa_wcomb_kernel(
        const float* __restrict__ wconv, const float* __restrict__ wproj,
        float* __restrict__ wcomb) {
    const int gid = blockIdx.x * 256 + threadIdx.x;      // 128*32
    const int a = gid >> 5, k = gid & 31;
    float acc = 0.f;
    if (k < KSZ_)
        for (int f = 0; f < NFILT_; ++f)
            acc += wproj[a * NFILT_ + f] * wconv[f * KSZ_ + k];
    wcomb[gid] = acc;
}

// ---- pipelined, double-buffered WMMA scores kernel --------------------------
// 512 threads / 16 waves: wave = (t-subtile 0..7, a-half 0..1).

__device__ __forceinline__ void load_chunk_regs(
        int chunk, int b, int t0, int tid,
        const float* __restrict__ keys, const float* __restrict__ wkey,
        const float* __restrict__ awc,  const float* __restrict__ wcomb,
        float* ra, float* rb) {
    if (chunk < 16) {
#pragma unroll
        for (int i = 0; i < 2; ++i) {
            const int idx = i * 512 + tid;
            const int r = idx >> 3, c4 = (idx & 7) << 2;
            const float4 va = *(const float4*)&keys[(size_t)(b * T_ + t0 + r) * ENC_ + chunk * 32 + c4];
            const float4 vb = *(const float4*)&wkey[r * ENC_ + chunk * 32 + c4];
            ra[i*4+0] = va.x; ra[i*4+1] = va.y; ra[i*4+2] = va.z; ra[i*4+3] = va.w;
            rb[i*4+0] = vb.x; rb[i*4+1] = vb.y; rb[i*4+2] = vb.z; rb[i*4+3] = vb.w;
        }
    } else {
#pragma unroll
        for (int i = 0; i < 2; ++i) {
            const int idx = i * 512 + tid;
            const int r = idx >> 3, c4 = (idx & 7) << 2;
            const float4 vb = *(const float4*)&wcomb[r * 32 + c4];
            rb[i*4+0] = vb.x; rb[i*4+1] = vb.y; rb[i*4+2] = vb.z; rb[i*4+3] = vb.w;
#pragma unroll
            for (int q = 0; q < 4; ++q) {
                const int c = c4 + q;
                const int pos = t0 + r + c - PAD_;
                ra[i*4+q] = (c < KSZ_ && pos >= 0 && pos < T_) ? awc[b * T_ + pos] : 0.f;
            }
        }
    }
}

__device__ __forceinline__ void store_chunk_lds(
        unsigned short* lA, unsigned short* lB, int tid,
        const float* ra, const float* rb) {
#pragma unroll
    for (int i = 0; i < 2; ++i) {
        const int idx = i * 512 + tid;
        const int r = idx >> 3, c4 = (idx & 7) << 2;
        uint2 pa, pb;
        pa.x = (unsigned)f2bf(ra[i*4+0]) | ((unsigned)f2bf(ra[i*4+1]) << 16);
        pa.y = (unsigned)f2bf(ra[i*4+2]) | ((unsigned)f2bf(ra[i*4+3]) << 16);
        pb.x = (unsigned)f2bf(rb[i*4+0]) | ((unsigned)f2bf(rb[i*4+1]) << 16);
        pb.y = (unsigned)f2bf(rb[i*4+2]) | ((unsigned)f2bf(rb[i*4+3]) << 16);
        *(uint2*)&lA[r * LDS_STRIDE + c4] = pa;
        *(uint2*)&lB[r * LDS_STRIDE + c4] = pb;
    }
}

__global__ __launch_bounds__(512) void lsa_scores_kernel(
        const float* __restrict__ keys, const float* __restrict__ awc,
        const float* __restrict__ wkey, const float* __restrict__ wcomb,
        const float* __restrict__ qproj, const float* __restrict__ wattn,
        float* __restrict__ scores) {
    __shared__ unsigned short sA[2][128 * LDS_STRIDE];
    __shared__ unsigned short sB[2][128 * LDS_STRIDE];
    __shared__ float sred[128];
    const int tid = threadIdx.x, lane = tid & 31, wave = tid >> 5;
    const int b = blockIdx.y, t0 = blockIdx.x * 128;
    const int tsub = (wave >> 1) << 4;   // t-subtile origin: 0..112
    const int ahalf = wave & 1;          // a-tiles [ahalf*4, ahalf*4+4)

    v8f acc[4];
#pragma unroll
    for (int j = 0; j < 4; ++j)
#pragma unroll
        for (int r = 0; r < 8; ++r) acc[j][r] = 0.f;

    float ra[8], rb[8];
    load_chunk_regs(0, b, t0, tid, keys, wkey, awc, wcomb, ra, rb);

    for (int c = 0; c < 17; ++c) {
        unsigned short* lA = sA[c & 1];
        unsigned short* lB = sB[c & 1];
        store_chunk_lds(lA, lB, tid, ra, rb);
        if (c + 1 < 17)   // next chunk's global loads stay in flight across barrier
            load_chunk_regs(c + 1, b, t0, tid, keys, wkey, awc, wcomb, ra, rb);
        LDS_BARRIER();    // s_wait_dscnt 0 + split barrier: no loadcnt flush
        const v16bf aF  = load_a_frag(lA, tsub, lane);
        const v16bf bF0 = load_b_frag(lB, (ahalf * 4 + 0) * 16, lane);
        const v16bf bF1 = load_b_frag(lB, (ahalf * 4 + 1) * 16, lane);
        const v16bf bF2 = load_b_frag(lB, (ahalf * 4 + 2) * 16, lane);
        const v16bf bF3 = load_b_frag(lB, (ahalf * 4 + 3) * 16, lane);
        SCHED_FENCE();    // all fragment ds_loads issued before the WMMA chain
        acc[0] = __builtin_amdgcn_wmma_f32_16x16x32_bf16(
            false, aF, false, bF0, (short)0, acc[0], false, false);
        acc[1] = __builtin_amdgcn_wmma_f32_16x16x32_bf16(
            false, aF, false, bF1, (short)0, acc[1], false, false);
        acc[2] = __builtin_amdgcn_wmma_f32_16x16x32_bf16(
            false, aF, false, bF2, (short)0, acc[2], false, false);
        acc[3] = __builtin_amdgcn_wmma_f32_16x16x32_bf16(
            false, aF, false, bF3, (short)0, acc[3], false, false);
        SCHED_FENCE();
        // Buffer reuse across iterations is safe: each wave executes
        // s_wait_dscnt 0 before signaling the barrier, so all of its fragment
        // reads of buffer (c-2) completed before any wave restores it at c.
    }

    // Epilogue: C layout: VGPR r, lane L -> N = L%16, M = r + 8*(L/16)
    const int half = (lane >> 4) & 1, mrow = lane & 15;
    float pr[8];
#pragma unroll
    for (int r = 0; r < 8; ++r) pr[r] = 0.f;
#pragma unroll
    for (int j = 0; j < 4; ++j) {
        const int a = (ahalf * 4 + j) * 16 + mrow;
        const float q  = qproj[b * ATTN_ + a];
        const float wv = wattn[a];
#pragma unroll
        for (int r = 0; r < 8; ++r)
            pr[r] += wv * tanhf(acc[j][r] + q);
    }
#pragma unroll
    for (int r = 0; r < 8; ++r) {
        pr[r] += __shfl_xor(pr[r], 8, 32);
        pr[r] += __shfl_xor(pr[r], 4, 32);
        pr[r] += __shfl_xor(pr[r], 2, 32);
        pr[r] += __shfl_xor(pr[r], 1, 32);
    }
    if (ahalf == 0 && mrow == 0) {
#pragma unroll
        for (int r = 0; r < 8; ++r)
            sred[tsub + half * 8 + r] = pr[r];
    }
    __syncthreads();
    if (ahalf == 1 && mrow == 0) {
#pragma unroll
        for (int r = 0; r < 8; ++r)
            scores[b * T_ + t0 + tsub + half * 8 + r] =
                pr[r] + sred[tsub + half * 8 + r];
    }
}

// ---- softmax over T ---------------------------------------------------------

__global__ __launch_bounds__(256) void lsa_softmax_kernel(
        const float* __restrict__ scores, const unsigned char* __restrict__ mask,
        float* __restrict__ attn) {
    const int b = blockIdx.x, tid = threadIdx.x;
    __shared__ float red[256];
    float vals[4];
    float m = -INFINITY;
#pragma unroll
    for (int i = 0; i < 4; ++i) {
        const int t = tid + i * 256;
        float s = scores[b * T_ + t];
        if (mask[b * T_ + t]) s = -INFINITY;
        vals[i] = s;
        m = fmaxf(m, s);
    }
    red[tid] = m; __syncthreads();
    for (int s = 128; s > 0; s >>= 1) {
        if (tid < s) red[tid] = fmaxf(red[tid], red[tid + s]);
        __syncthreads();
    }
    m = red[0]; __syncthreads();
    float sum = 0.f;
#pragma unroll
    for (int i = 0; i < 4; ++i) { vals[i] = __expf(vals[i] - m); sum += vals[i]; }
    red[tid] = sum; __syncthreads();
    for (int s = 128; s > 0; s >>= 1) {
        if (tid < s) red[tid] += red[tid + s];
        __syncthreads();
    }
    const float inv = 1.f / red[0];
#pragma unroll
    for (int i = 0; i < 4; ++i)
        attn[b * T_ + tid + i * 256] = vals[i] * inv;
}

// ---- ctx_v partials: slice t across 8 blocks/b, float4 value loads ----------

__global__ __launch_bounds__(128) void lsa_ctxv_part_kernel(
        const float* __restrict__ values, const float* __restrict__ attn,
        float* __restrict__ part) {
    const int b = blockIdx.y, slice = blockIdx.x, tid = threadIdx.x;  // tid: e/4
    const float4* vp = (const float4*)values;
    const int tbase = slice * 128;
    float4 acc; acc.x = acc.y = acc.z = acc.w = 0.f;
#pragma unroll 4
    for (int i = 0; i < 128; ++i) {
        const int t = tbase + i;
        const float a = attn[b * T_ + t];
        const float4 v = vp[(size_t)(b * T_ + t) * (ENC_ / 4) + tid];
        acc.x += a * v.x; acc.y += a * v.y; acc.z += a * v.z; acc.w += a * v.w;
    }
    float4* pp = (float4*)part;
    pp[((size_t)slice * B_ + b) * (ENC_ / 4) + tid] = acc;
}

__global__ __launch_bounds__(256) void lsa_ctxv_reduce_kernel(
        const float* __restrict__ part, float* __restrict__ ctxv) {
    const int idx = blockIdx.x * 256 + threadIdx.x;     // b*ENC + e, 32768 total
    float acc = 0.f;
#pragma unroll
    for (int s = 0; s < 8; ++s)
        acc += part[s * (B_ * ENC_) + idx];
    ctxv[idx] = acc;
}

// ---- context[b,f] = sum_e ctx_v[b,e]*W_value[f,e] ---------------------------

__global__ __launch_bounds__(256) void lsa_context_kernel(
        const float* __restrict__ ctxv, const float* __restrict__ wvalue,
        float* __restrict__ ctx) {
    const int b = blockIdx.x, tid = threadIdx.x;
#pragma unroll
    for (int p = 0; p < 2; ++p) {
        const int f = tid + p * 256;
        float acc = 0.f;
        for (int e = 0; e < ENC_; ++e)
            acc += ctxv[b * ENC_ + e] * wvalue[f * ENC_ + e];
        ctx[b * ENC_ + f] = acc;
    }
}

extern "C" void kernel_launch(void* const* d_in, const int* in_sizes, int n_in,
                              void* d_out, int out_size, void* d_ws, size_t ws_size,
                              hipStream_t stream) {
    const float* query  = (const float*)d_in[0];
    const float* keys   = (const float*)d_in[1];
    const float* values = (const float*)d_in[2];
    const float* awc    = (const float*)d_in[3];
    const unsigned char* mask = (const unsigned char*)d_in[4];
    const float* wconv  = (const float*)d_in[5];
    const float* wproj  = (const float*)d_in[6];
    const float* wq     = (const float*)d_in[7];
    const float* wkey   = (const float*)d_in[8];
    const float* wvalue = (const float*)d_in[9];
    const float* wattn  = (const float*)d_in[10];

    float* out  = (float*)d_out;
    float* ctx  = out;                 // (B, ENC) = 32768 floats
    float* attn = out + B_ * ENC_;     // (B, T)   = 65536 floats

    float* ws     = (float*)d_ws;
    float* qproj  = ws;                                  // 8192
    float* wcomb  = ws + 8192;                           // 4096
    float* scores = ws + 8192 + 4096;                    // 65536
    float* ctxv   = ws + 8192 + 4096 + 65536;            // 32768
    float* part   = ws + 8192 + 4096 + 65536 + 32768;    // 8*64*512 = 262144

    lsa_qproj_kernel<<<32, 256, 0, stream>>>(query, wq, qproj);
    lsa_wcomb_kernel<<<16, 256, 0, stream>>>(wconv, wproj, wcomb);
    lsa_scores_kernel<<<dim3(T_ / 128, B_), 512, 0, stream>>>(
        keys, awc, wkey, wcomb, qproj, wattn, scores);
    lsa_softmax_kernel<<<B_, 256, 0, stream>>>(scores, mask, attn);
    lsa_ctxv_part_kernel<<<dim3(8, B_), 128, 0, stream>>>(values, attn, part);
    lsa_ctxv_reduce_kernel<<<B_ * ENC_ / 256, 256, 0, stream>>>(part, ctxv);
    lsa_context_kernel<<<B_, 256, 0, stream>>>(ctxv, wvalue, ctx);
}
